// GraphTransformerBlock_79688823210238
// MI455X (gfx1250) — matrix operand order repaired and checked by reading, not compiled, not run on hardware
//
#include <hip/hip_runtime.h>
#include <cmath>

// ---------------------------------------------------------------------------
// GraphTransformerBlock for MI455X (gfx1250, wave32, WMMA bf16 path)
// N=50000 HID=256 HEADS=8 HD=32 K=16 ; dst = repeat(arange(N),16) (contiguous)
// GEMMs: 1x4 register-blocked WMMA (one A fragment reused over 4 B tiles).
// ---------------------------------------------------------------------------

typedef __attribute__((ext_vector_type(16))) __bf16 v16bf;
typedef __attribute__((ext_vector_type(8)))  __bf16 v8bf;
typedef __attribute__((ext_vector_type(8)))  float  v8f;
typedef __attribute__((ext_vector_type(4)))  float  v4f;

#define HID   256
#define HEADS 8
#define HD    32
#define NBRS  16
#define SCALE 0.17677669529663687f   // 32^-0.5
#define LN_EPS 1e-5f
#define APAD  8                      // LDS row skew (elems) to avoid bank conflicts

__device__ __forceinline__ float wave_sum(float v) {
    #pragma unroll
    for (int off = 16; off; off >>= 1) v += __shfl_xor(v, off, 32);
    return v;
}
__device__ __forceinline__ float wave_max(float v) {
    #pragma unroll
    for (int off = 16; off; off >>= 1) v = fmaxf(v, __shfl_xor(v, off, 32));
    return v;
}

// ---------------------------------------------------------------------------
// Pack fp32 weight (Kdim x Ndim row-major) into WMMA bf16 B-operand layout,
// kt-major tile order: tile = kt*Ntiles + nt, so all col-tiles of one k-step
// are contiguous.  Within a tile (512 bf16): lane l holds column
// n = nt*16 + (l&15), elements t=0..15 -> K = kt*32 + (l>>4)*16 + t.
// One wave per tile.
// ---------------------------------------------------------------------------
__global__ __launch_bounds__(32) void pack_weight_bf16(
    const float* __restrict__ W, __bf16* __restrict__ P, int Kdim, int Ndim)
{
    int tile   = blockIdx.x;               // tile = kt*Ntiles + nt
    int Ntiles = Ndim >> 4;
    int kt     = tile / Ntiles;
    int nt     = tile - kt * Ntiles;
    int lane   = threadIdx.x & 31;
    int n      = nt * 16 + (lane & 15);
    int kbase  = kt * 32 + (lane >> 4) * 16;
    __bf16* dst = P + (size_t)tile * 512 + lane * 16;
    #pragma unroll
    for (int t = 0; t < 16; ++t)
        dst[t] = (__bf16)W[(size_t)(kbase + t) * Ndim + n];
}

// ---------------------------------------------------------------------------
// LayerNorm(row of 256 fp32) -> bf16.  One wave per row, 8 elems per lane.
// ---------------------------------------------------------------------------
__global__ __launch_bounds__(256) void ln_to_bf16(
    const float* __restrict__ X, const float* __restrict__ g,
    const float* __restrict__ b, __bf16* __restrict__ H, int n)
{
    int row  = blockIdx.x * 8 + (threadIdx.x >> 5);
    int lane = threadIdx.x & 31;
    if (row >= n) return;
    const float* xr = X + (size_t)row * HID;
    v4f x0 = *(const v4f*)(xr + lane * 8);
    v4f x1 = *(const v4f*)(xr + lane * 8 + 4);
    float s = 0.f, s2 = 0.f;
    #pragma unroll
    for (int i = 0; i < 4; ++i) {
        s  += x0[i] + x1[i];
        s2 += x0[i] * x0[i] + x1[i] * x1[i];
    }
    s = wave_sum(s); s2 = wave_sum(s2);
    float mean = s * (1.f / HID);
    float var  = s2 * (1.f / HID) - mean * mean;
    float inv  = rsqrtf(var + LN_EPS);
    v8bf o;
    #pragma unroll
    for (int i = 0; i < 4; ++i) {
        o[i]     = (__bf16)((x0[i] - mean) * inv * g[lane * 8 + i]     + b[lane * 8 + i]);
        o[i + 4] = (__bf16)((x1[i] - mean) * inv * g[lane * 8 + 4 + i] + b[lane * 8 + 4 + i]);
    }
    *(v8bf*)(H + (size_t)row * HID + lane * 8) = o;
}

// ---------------------------------------------------------------------------
// bf16 WMMA GEMM, 1x4 register blocking.
//   block = 256 threads = 8 waves, covering ALL Ndim columns:
//     CG col-groups (64 cols each) x (8/CG) row-groups (16 rows each)
//   rowsPerBlock = 128/CG ; A rows staged in LDS with APAD skew.
//   Wave: one A fragment per k-step reused across 4 B fragments -> 4 WMMA.
// Epilogue: +bias[n], +residual[m,n], relu, fp32 and/or bf16 store.
// ---------------------------------------------------------------------------
__global__ __launch_bounds__(256) void gemm_bf16_wmma(
    const __bf16* __restrict__ A, int Kdim,
    const __bf16* __restrict__ Bp, int Ktiles, int Ntiles, int CG,
    const float* __restrict__ bias, const float* __restrict__ residual,
    float* __restrict__ outF, __bf16* __restrict__ outB,
    int ldc, int relu, int M)
{
    __shared__ __bf16 Atile[32 * (512 + APAD)];   // up to 32 rows x 520 (33.3 KB)
    const int pitch   = Kdim + APAD;
    const int RP      = 128 / CG;                  // rows per block (16 or 32)
    const int rowBase = blockIdx.x * RP;
    const int tid     = threadIdx.x;

    // cooperative A-tile load (RP x Kdim bf16), 16B chunks, clamped rows
    {
        const int cpr    = Kdim >> 3;              // chunks per row
        const int chunks = RP * cpr;
        for (int c = tid; c < chunks; c += 256) {
            int r  = c / cpr;
            int k  = (c - r * cpr) << 3;
            int gr = rowBase + r;
            if (gr >= M) gr = M - 1;               // clamp (dup row, never stored)
            *(v8bf*)(&Atile[r * pitch + k]) =
                *(const v8bf*)(&A[(size_t)gr * Kdim + k]);
        }
    }
    __syncthreads();

    const int wave = tid >> 5;
    const int lane = tid & 31;
    const int rg   = wave / CG;                    // row-group within block
    const int cg   = wave - rg * CG;               // col-group (64 columns)
    const int half = lane >> 4;
    const int mrow = lane & 15;
    const int rowT = rg * 16;                      // LDS row offset

    v8f acc0 = {}, acc1 = {}, acc2 = {}, acc3 = {};
    const __bf16* brow = Bp + (size_t)cg * 4 * 512 + lane * 16;
    const size_t  bstep = (size_t)Ntiles * 512;    // advance one k-step

    for (int kt = 0; kt < Ktiles; ++kt) {
        __builtin_prefetch(brow + bstep, 0, 1);    // next k-step's B tiles
        // A fragment (shared across the 4 WMMAs below)
        const __bf16* ap = &Atile[(rowT + mrow) * pitch + kt * 32 + half * 16];
        v8bf a0 = *(const v8bf*)(ap);
        v8bf a1 = *(const v8bf*)(ap + 8);
        v16bf af = __builtin_shufflevector(a0, a1, 0,1,2,3,4,5,6,7,8,9,10,11,12,13,14,15);
        // 4 B fragments: contiguous tiles (kt-major packing)
        #pragma unroll
        for (int j = 0; j < 4; ++j) {
            v8bf b0 = *(const v8bf*)(brow + j * 512);
            v8bf b1 = *(const v8bf*)(brow + j * 512 + 8);
            v16bf bfr = __builtin_shufflevector(b0, b1, 0,1,2,3,4,5,6,7,8,9,10,11,12,13,14,15);
            v8f* accp = (j == 0) ? &acc0 : (j == 1) ? &acc1 : (j == 2) ? &acc2 : &acc3;
            *accp = __builtin_amdgcn_wmma_f32_16x16x32_bf16(
                false, af, false, bfr, (short)0, *accp, false, false);
        }
        brow += bstep;
    }

    // Epilogue.  D layout: VGPR r -> (M = r + 8*half, N = lane&15).
    #pragma unroll
    for (int j = 0; j < 4; ++j) {
        const v8f acc = (j == 0) ? acc0 : (j == 1) ? acc1 : (j == 2) ? acc2 : acc3;
        const int gn  = (cg * 4 + j) * 16 + mrow;
        const float bs = bias ? bias[gn] : 0.f;
        #pragma unroll
        for (int r = 0; r < 8; ++r) {
            int gm = rowBase + rowT + r + half * 8;
            if (gm >= M) continue;
            float v = acc[r] + bs;
            if (residual) v += residual[(size_t)gm * ldc + gn];
            if (relu)     v  = v > 0.f ? v : 0.f;
            size_t oi = (size_t)gm * ldc + gn;
            if (outF) outF[oi] = v;
            if (outB) outB[oi] = (__bf16)v;
        }
    }
}

// ---------------------------------------------------------------------------
// Attention: one wave per (node, head).  dst is contiguous groups of 16.
// lanes 0..15 own edges (src idx + geometric bias); dot via wave reduction;
// 16-wide softmax; coalesced p-weighted V gather -> bf16 agg.
// ---------------------------------------------------------------------------
__global__ __launch_bounds__(256) void attn_kernel(
    const float* __restrict__ Q, const float* __restrict__ Km,
    const float* __restrict__ V, const int* __restrict__ src,
    const float* __restrict__ pos, const float* __restrict__ edge_w,
    const float* __restrict__ edge_b, __bf16* __restrict__ aggB, int n)
{
    int node = blockIdx.x;
    if (node >= n) return;
    int head = threadIdx.x >> 5;
    int lane = threadIdx.x & 31;

    int   sidx = 0;
    float bias = 0.f;
    if (lane < NBRS) {
        sidx = src[node * NBRS + lane];
        float r0 = pos[sidx * 3 + 0] - pos[node * 3 + 0];
        float r1 = pos[sidx * 3 + 1] - pos[node * 3 + 1];
        float r2 = pos[sidx * 3 + 2] - pos[node * 3 + 2];
        float dist = sqrtf(r0 * r0 + r1 * r1 + r2 * r2);
        bias = r0 * edge_w[0 * HEADS + head] + r1 * edge_w[1 * HEADS + head]
             + r2 * edge_w[2 * HEADS + head] + dist * edge_w[3 * HEADS + head]
             + edge_b[head];
    }

    float qv = Q[(size_t)node * HID + head * HD + lane];
    float a  = -INFINITY;
    #pragma unroll
    for (int e = 0; e < NBRS; ++e) {
        int   se = __shfl(sidx, e, 32);
        float kv = Km[(size_t)se * HID + head * HD + lane];
        float d  = wave_sum(qv * kv);
        if (lane == e) a = d * SCALE + bias;
    }
    float m   = wave_max(a);                       // lanes>=16 hold -inf
    float ex  = (lane < NBRS) ? __expf(a - m) : 0.f;
    float sum = fmaxf(wave_sum(ex), 1e-6f);
    float p   = ex / sum;

    float acc = 0.f;
    #pragma unroll
    for (int e = 0; e < NBRS; ++e) {
        float pe = __shfl(p, e, 32);
        int   se = __shfl(sidx, e, 32);
        acc += pe * V[(size_t)se * HID + head * HD + lane];
    }
    aggB[(size_t)node * HID + head * HD + lane] = (__bf16)acc;
}

// ---------------------------------------------------------------------------
extern "C" void kernel_launch(void* const* d_in, const int* in_sizes, int n_in,
                              void* d_out, int out_size, void* d_ws, size_t ws_size,
                              hipStream_t stream)
{
    const float* x      = (const float*)d_in[0];
    const float* pos    = (const float*)d_in[1];
    const int*   src    = (const int*)  d_in[2];
    /* d_in[3] = dst : structurally repeat(arange(n),16) -> unused */
    const float* ln1_g  = (const float*)d_in[4];
    const float* ln1_b  = (const float*)d_in[5];
    const float* q_w    = (const float*)d_in[6];
    const float* k_w    = (const float*)d_in[7];
    const float* v_w    = (const float*)d_in[8];
    const float* edge_w = (const float*)d_in[9];
    const float* edge_b = (const float*)d_in[10];
    const float* out_w  = (const float*)d_in[11];
    const float* out_b  = (const float*)d_in[12];
    const float* ln2_g  = (const float*)d_in[13];
    const float* ln2_b  = (const float*)d_in[14];
    const float* ffn_w1 = (const float*)d_in[15];
    const float* ffn_b1 = (const float*)d_in[16];
    const float* ffn_w2 = (const float*)d_in[17];
    const float* ffn_b2 = (const float*)d_in[18];
    float* out = (float*)d_out;

    const int n = in_sizes[0] / HID;         // 50000

    // ---- workspace carve-up (256B aligned) --------------------------------
    char* wp = (char*)d_ws;
    auto carve = [&](size_t bytes) {
        char* p = wp;
        wp += (bytes + 255) & ~(size_t)255;
        return p;
    };
    __bf16* qwP  = (__bf16*)carve((size_t)HID * HID * 2);
    __bf16* kwP  = (__bf16*)carve((size_t)HID * HID * 2);
    __bf16* vwP  = (__bf16*)carve((size_t)HID * HID * 2);
    __bf16* owP  = (__bf16*)carve((size_t)HID * HID * 2);
    __bf16* w1P  = (__bf16*)carve((size_t)HID * 2 * HID * 2);
    __bf16* w2P  = (__bf16*)carve((size_t)2 * HID * HID * 2);
    __bf16* hbf  = (__bf16*)carve((size_t)n * HID * 2);
    float*  Qf   = (float*) carve((size_t)n * HID * 4);
    float*  Kf   = (float*) carve((size_t)n * HID * 4);
    float*  Vf   = (float*) carve((size_t)n * HID * 4);
    __bf16* agg  = (__bf16*)carve((size_t)n * HID * 2);
    float*  x1   = (float*) carve((size_t)n * HID * 4);
    __bf16* h2bf = (__bf16*)carve((size_t)n * HID * 2);
    __bf16* f1bf = (__bf16*)carve((size_t)n * 2 * HID * 2);

    const int Mb32 = (n + 31) / 32;          // blocks of 32 rows (CG=4)
    const int Mb16 = (n + 15) / 16;          // blocks of 16 rows (CG=8)

    // ---- 1. pack weights into WMMA B layout (kt-major tiles) --------------
    pack_weight_bf16<<<dim3(128), 32, 0, stream>>>(q_w,    qwP, HID, HID);
    pack_weight_bf16<<<dim3(128), 32, 0, stream>>>(k_w,    kwP, HID, HID);
    pack_weight_bf16<<<dim3(128), 32, 0, stream>>>(v_w,    vwP, HID, HID);
    pack_weight_bf16<<<dim3(128), 32, 0, stream>>>(out_w,  owP, HID, HID);
    pack_weight_bf16<<<dim3(256), 32, 0, stream>>>(ffn_w1, w1P, HID, 2 * HID);
    pack_weight_bf16<<<dim3(256), 32, 0, stream>>>(ffn_w2, w2P, 2 * HID, HID);

    // ---- 2. h = LN1(x) -> bf16 -------------------------------------------
    ln_to_bf16<<<dim3((n + 7) / 8), 256, 0, stream>>>(x, ln1_g, ln1_b, hbf, n);

    // ---- 3. Q/K/V = h @ W   (Kdim=256, N=256, CG=4) -----------------------
    gemm_bf16_wmma<<<dim3(Mb32), 256, 0, stream>>>(hbf, HID, qwP, 8, 16, 4,
        nullptr, nullptr, Qf, nullptr, HID, 0, n);
    gemm_bf16_wmma<<<dim3(Mb32), 256, 0, stream>>>(hbf, HID, kwP, 8, 16, 4,
        nullptr, nullptr, Kf, nullptr, HID, 0, n);
    gemm_bf16_wmma<<<dim3(Mb32), 256, 0, stream>>>(hbf, HID, vwP, 8, 16, 4,
        nullptr, nullptr, Vf, nullptr, HID, 0, n);

    // ---- 4. neighbor attention + softmax + aggregation -------------------
    attn_kernel<<<dim3(n), 256, 0, stream>>>(Qf, Kf, Vf, src, pos,
                                             edge_w, edge_b, agg, n);

    // ---- 5. x1 = x + agg @ out_w + out_b ---------------------------------
    gemm_bf16_wmma<<<dim3(Mb32), 256, 0, stream>>>(agg, HID, owP, 8, 16, 4,
        out_b, x, x1, nullptr, HID, 0, n);

    // ---- 6. h2 = LN2(x1) -> bf16 -----------------------------------------
    ln_to_bf16<<<dim3((n + 7) / 8), 256, 0, stream>>>(x1, ln2_g, ln2_b, h2bf, n);

    // ---- 7. f1 = relu(h2 @ w1 + b1) -> bf16  (N=512, CG=8) ----------------
    gemm_bf16_wmma<<<dim3(Mb16), 256, 0, stream>>>(h2bf, HID, w1P, 8, 32, 8,
        ffn_b1, nullptr, nullptr, f1bf, 2 * HID, 1, n);

    // ---- 8. out = x1 + f1 @ w2 + b2  (Kdim=512, N=256, CG=4) --------------
    gemm_bf16_wmma<<<dim3(Mb32), 256, 0, stream>>>(f1bf, 2 * HID, w2P, 16, 16, 4,
        ffn_b2, x1, out, nullptr, HID, 0, n);
}
